// SzDense_18107582120284
// MI455X (gfx1250) — matrix operand order repaired and checked by reading, not compiled, not run on hardware
//
#include <hip/hip_runtime.h>

// out[M,N] = x[M,K] @ (kernel*window)[K,N] + bias[N], all 4096, f32 reference.
#define MM  4096
#define KK  4096
#define NN  4096

#define BM 128
#define BN 128
#define BK 32
#define APAD 8
#define BPAD 8
#define THREADS 256

typedef __attribute__((ext_vector_type(16))) __bf16 v16bf;
typedef __attribute__((ext_vector_type(8)))  float  v8f;
typedef unsigned int u32x4 __attribute__((ext_vector_type(4)));
typedef int          i32x4 __attribute__((ext_vector_type(4)));
typedef int          i32x8 __attribute__((ext_vector_type(8)));

union FragU { uint4 q[2]; v16bf v; };

__device__ __forceinline__ unsigned short f32_to_bf16_rne(float f) {
    unsigned u = __float_as_uint(f);
    unsigned r = 0x7FFFu + ((u >> 16) & 1u);
    return (unsigned short)((u + r) >> 16);
}
__device__ __forceinline__ float bf16_hi_f32(unsigned short h) {
    return __uint_as_float(((unsigned)h) << 16);
}
__device__ __forceinline__ void split_bf16x2(float f, unsigned short& hi, unsigned short& lo) {
    hi = f32_to_bf16_rne(f);
    lo = f32_to_bf16_rne(f - bf16_hi_f32(hi));
}

// ---------------------------------------------------------------------------
// TDM: 2D tile load global->LDS. Descriptor per CDNA5 ISA ch.8 (D# groups 0/1).
// tile_w elements per row (2B each), tile_h rows, tensor 4096x4096, row stride
// 4096 elements. pad_interval/pad_amount insert the LDS anti-bank-conflict pad.
// ---------------------------------------------------------------------------
__device__ __forceinline__ void tdm_load_2d(unsigned lds_off, const void* gptr,
                                            unsigned tile_w, unsigned tile_h,
                                            unsigned pad_interval, unsigned pad_amount) {
    unsigned long long ga = (unsigned long long)(uintptr_t)gptr;
    u32x4 g0;
    g0[0] = 1u;                                       // count=1, user-mode, no gather
    g0[1] = lds_off;                                  // LDS byte address
    g0[2] = (unsigned)(ga & 0xFFFFFFFFu);             // global_addr[31:0]
    g0[3] = (unsigned)((ga >> 32) & 0x01FFFFFFu) | (2u << 30); // addr[56:32] | type=2
    i32x8 g1;
    g1[0] = (int)((1u << 16) |                        // data_size = 2 bytes
                  (1u << 20) |                        // pad_enable
                  (pad_interval << 22) | (pad_amount << 25));
    g1[1] = (int)(4096u << 16);                       // tensor_dim0[15:0] in [63:48]
    g1[2] = (int)(4096u << 16);                       // dim0 hi=0 | tensor_dim1[15:0]
    g1[3] = (int)(tile_w << 16);                      // dim1 hi=0 | tile_dim0
    g1[4] = (int)tile_h;                              // tile_dim1 | tile_dim2=0
    g1[5] = 4096;                                     // tensor_dim0_stride[31:0]
    g1[6] = 0;                                        // stride0 hi | dim1_stride lo (2D: unused)
    g1[7] = 0;
    i32x4 z4 = {0, 0, 0, 0};
#if __has_builtin(__builtin_amdgcn_tensor_load_to_lds)
  #if defined(__clang_major__) && (__clang_major__ >= 23)
    i32x8 z8 = {0, 0, 0, 0, 0, 0, 0, 0};
    __builtin_amdgcn_tensor_load_to_lds(g0, g1, z4, z4, z8, 0);
  #else
    __builtin_amdgcn_tensor_load_to_lds(g0, g1, z4, z4, 0);
  #endif
#else
    asm volatile("tensor_load_to_lds %0, %1, %2, %3"
                 :: "s"(g0), "s"(g1), "s"(z4), "s"(z4) : "memory");
#endif
}

// ---------------------------------------------------------------------------
// Pre-pass: split x and (kernel*window) into bf16 hi/lo planes, once per element.
// ---------------------------------------------------------------------------
__global__ __launch_bounds__(256)
void szdense_split_prepass(const float* __restrict__ X,
                           const float* __restrict__ Kw,
                           const float* __restrict__ Wn,
                           unsigned short* __restrict__ Xhi, unsigned short* __restrict__ Xlo,
                           unsigned short* __restrict__ Phi, unsigned short* __restrict__ Plo) {
    size_t i = ((size_t)blockIdx.x * 256 + threadIdx.x) * 4;
    float4 xv = *(const float4*)&X[i];
    ushort4 h4, l4;
    split_bf16x2(xv.x, h4.x, l4.x); split_bf16x2(xv.y, h4.y, l4.y);
    split_bf16x2(xv.z, h4.z, l4.z); split_bf16x2(xv.w, h4.w, l4.w);
    *(ushort4*)&Xhi[i] = h4;
    *(ushort4*)&Xlo[i] = l4;

    float4 kv = *(const float4*)&Kw[i];
    float4 wv = *(const float4*)&Wn[i];
    float4 p; p.x = kv.x * wv.x; p.y = kv.y * wv.y; p.z = kv.z * wv.z; p.w = kv.w * wv.w;
    split_bf16x2(p.x, h4.x, l4.x); split_bf16x2(p.y, h4.y, l4.y);
    split_bf16x2(p.z, h4.z, l4.z); split_bf16x2(p.w, h4.w, l4.w);
    *(ushort4*)&Phi[i] = h4;
    *(ushort4*)&Plo[i] = l4;
}

// ---------------------------------------------------------------------------
// GEMM with TDM double-buffered tile fetch + bf16 split-emulation WMMA.
// ---------------------------------------------------------------------------
__global__ __launch_bounds__(THREADS)
void szdense_wmma_tdm(const unsigned short* __restrict__ Xhi,
                      const unsigned short* __restrict__ Xlo,
                      const unsigned short* __restrict__ Phi,
                      const unsigned short* __restrict__ Plo,
                      const float* __restrict__ Bias,
                      float* __restrict__ Out) {
    // [stage][hi/lo] tiles; row pads match the TDM descriptor padding.
    __shared__ alignas(16) unsigned short As[2][2][BM][BK + APAD];
    __shared__ alignas(16) unsigned short Bs[2][2][BK][BN + BPAD];

    const int tid  = threadIdx.x;
    const int lane = tid & 31;
    const int wave = tid >> 5;
    const int wm   = wave >> 2;            // 0..1
    const int wn   = wave & 3;             // 0..3

    const int nBase = blockIdx.x * BN;
    const int mBase = blockIdx.y * BM;

    const v8f vzero = {0.f,0.f,0.f,0.f,0.f,0.f,0.f,0.f};
    v8f acc[4][2];
    #pragma unroll
    for (int tm = 0; tm < 4; ++tm)
        #pragma unroll
        for (int tn = 0; tn < 2; ++tn)
            acc[tm][tn] = vzero;

    const int kc0 = (lane >> 4) << 3;      // A fragment K-chunk base per lane

    const int NIT = KK / BK;

    // Preload stage 0 (wave 0 drives the Tensor Data Mover; EXEC is ignored).
    if (wave == 0) {
        tdm_load_2d((unsigned)(uintptr_t)&As[0][0][0][0], Xhi + (size_t)mBase * KK, BK, BM, 3, 3);
        tdm_load_2d((unsigned)(uintptr_t)&As[0][1][0][0], Xlo + (size_t)mBase * KK, BK, BM, 3, 3);
        tdm_load_2d((unsigned)(uintptr_t)&Bs[0][0][0][0], Phi + nBase,              BN, BK, 5, 3);
        tdm_load_2d((unsigned)(uintptr_t)&Bs[0][1][0][0], Plo + nBase,              BN, BK, 5, 3);
    }

    for (int it = 0; it < NIT; ++it) {
        const int cur = it & 1;
        const int nxt = cur ^ 1;
        const int k0  = it * BK;

        // Everyone done reading buffer `nxt` (iteration it-1) before TDM rewrites it.
        __syncthreads();

        if (wave == 0) {
            if (it + 1 < NIT) {
                const int kn = k0 + BK;
                tdm_load_2d((unsigned)(uintptr_t)&As[nxt][0][0][0], Xhi + (size_t)mBase * KK + kn, BK, BM, 3, 3);
                tdm_load_2d((unsigned)(uintptr_t)&As[nxt][1][0][0], Xlo + (size_t)mBase * KK + kn, BK, BM, 3, 3);
                tdm_load_2d((unsigned)(uintptr_t)&Bs[nxt][0][0][0], Phi + (size_t)kn * NN + nBase, BN, BK, 5, 3);
                tdm_load_2d((unsigned)(uintptr_t)&Bs[nxt][1][0][0], Plo + (size_t)kn * NN + nBase, BN, BK, 5, 3);
                __builtin_amdgcn_s_wait_tensorcnt(4);   // stage `cur`'s 4 TDMs done; next 4 in flight
            } else {
                __builtin_amdgcn_s_wait_tensorcnt(0);
            }
        }
        __syncthreads();   // publish stage `cur` LDS to all waves

        // ---- fragment loads (ISA §7.12.2 layouts) ----
        FragU a[4][2];
        #pragma unroll
        for (int tm = 0; tm < 4; ++tm) {
            int m = wm * 64 + tm * 16 + (lane & 15);
            #pragma unroll
            for (int s = 0; s < 2; ++s) {
                a[tm][s].q[0] = *(const uint4*)&As[cur][s][m][kc0];
                a[tm][s].q[1] = *(const uint4*)&As[cur][s][m][kc0 + 16];
            }
        }
        FragU b[2][2];
        #pragma unroll
        for (int tn = 0; tn < 2; ++tn) {
            int n = wn * 32 + tn * 16;
            #pragma unroll
            for (int s = 0; s < 2; ++s) {
                for (int e = 0; e < 16; ++e)
                    ((unsigned short*)&b[tn][s])[e] = Bs[cur][s][kc0 + (e & 7) + ((e >> 3) << 4)][n + (lane & 15)];
            }
        }

        // ---- 3-product bf16 split emulation of the f32 GEMM ----
        #pragma unroll
        for (int tm = 0; tm < 4; ++tm) {
            #pragma unroll
            for (int tn = 0; tn < 2; ++tn) {
                acc[tm][tn] = __builtin_amdgcn_wmma_f32_16x16x32_bf16(
                    false, a[tm][0].v, false, b[tn][1].v, (short)0, acc[tm][tn], false, false);
                acc[tm][tn] = __builtin_amdgcn_wmma_f32_16x16x32_bf16(
                    false, a[tm][1].v, false, b[tn][0].v, (short)0, acc[tm][tn], false, false);
                acc[tm][tn] = __builtin_amdgcn_wmma_f32_16x16x32_bf16(
                    false, a[tm][0].v, false, b[tn][0].v, (short)0, acc[tm][tn], false, false);
            }
        }
    }

    // ---- epilogue: bias + store (C/D: M = r + 8*(lane>=16), N = lane&15) ----
    #pragma unroll
    for (int tn = 0; tn < 2; ++tn) {
        int n = nBase + wn * 32 + tn * 16 + (lane & 15);
        float bv = Bias[n];
        #pragma unroll
        for (int tm = 0; tm < 4; ++tm) {
            int m0 = mBase + wm * 64 + tm * 16 + ((lane >> 4) << 3);
            #pragma unroll
            for (int r = 0; r < 8; ++r)
                Out[(size_t)(m0 + r) * NN + n] = acc[tm][tn][r] + bv;
        }
    }
}

// ---------------------------------------------------------------------------
// Fallback (round-1 kernel): convert-in-kernel, used when d_ws is too small.
// ---------------------------------------------------------------------------
__global__ __launch_bounds__(THREADS)
void szdense_wmma_bf16x2(const float* __restrict__ X,
                         const float* __restrict__ Kw,
                         const float* __restrict__ Wn,
                         const float* __restrict__ Bias,
                         float* __restrict__ Out) {
    __shared__ alignas(16) unsigned short As[2][BM][BK + APAD];
    __shared__ alignas(16) unsigned short Bs[2][BK][BN + BPAD];

    const int tid  = threadIdx.x;
    const int lane = tid & 31;
    const int wave = tid >> 5;
    const int wm   = wave >> 2;
    const int wn   = wave & 3;

    const int nBase = blockIdx.x * BN;
    const int mBase = blockIdx.y * BM;

    const v8f vzero = {0.f,0.f,0.f,0.f,0.f,0.f,0.f,0.f};
    v8f acc[4][2];
    #pragma unroll
    for (int tm = 0; tm < 4; ++tm)
        #pragma unroll
        for (int tn = 0; tn < 2; ++tn)
            acc[tm][tn] = vzero;

    const int kc0 = (lane >> 4) << 3;

    for (int k0 = 0; k0 < KK; k0 += BK) {
        __syncthreads();
        #pragma unroll
        for (int i = 0; i < 4; ++i) {
            int idx4 = tid + i * THREADS;
            int row  = idx4 >> 3;
            int c4   = (idx4 & 7) << 2;
            const float4 v = *(const float4*)&X[(size_t)(mBase + row) * KK + k0 + c4];
            ushort4 h4, l4;
            split_bf16x2(v.x, h4.x, l4.x); split_bf16x2(v.y, h4.y, l4.y);
            split_bf16x2(v.z, h4.z, l4.z); split_bf16x2(v.w, h4.w, l4.w);
            *(ushort4*)&As[0][row][c4] = h4;
            *(ushort4*)&As[1][row][c4] = l4;
        }
        #pragma unroll
        for (int i = 0; i < 4; ++i) {
            int idx4 = tid + i * THREADS;
            int row  = idx4 >> 5;
            int c4   = (idx4 & 31) << 2;
            size_t g = (size_t)(k0 + row) * NN + nBase + c4;
            const float4 kv = *(const float4*)&Kw[g];
            const float4 wv = *(const float4*)&Wn[g];
            float4 p; p.x = kv.x * wv.x; p.y = kv.y * wv.y;
                      p.z = kv.z * wv.z; p.w = kv.w * wv.w;
            ushort4 h4, l4;
            split_bf16x2(p.x, h4.x, l4.x); split_bf16x2(p.y, h4.y, l4.y);
            split_bf16x2(p.z, h4.z, l4.z); split_bf16x2(p.w, h4.w, l4.w);
            *(ushort4*)&Bs[0][row][c4] = h4;
            *(ushort4*)&Bs[1][row][c4] = l4;
        }
        __syncthreads();

        FragU a[4][2];
        #pragma unroll
        for (int tm = 0; tm < 4; ++tm) {
            int m = wm * 64 + tm * 16 + (lane & 15);
            #pragma unroll
            for (int s = 0; s < 2; ++s) {
                a[tm][s].q[0] = *(const uint4*)&As[s][m][kc0];
                a[tm][s].q[1] = *(const uint4*)&As[s][m][kc0 + 16];
            }
        }
        FragU b[2][2];
        #pragma unroll
        for (int tn = 0; tn < 2; ++tn) {
            int n = wn * 32 + tn * 16;
            #pragma unroll
            for (int s = 0; s < 2; ++s) {
                for (int e = 0; e < 16; ++e)
                    ((unsigned short*)&b[tn][s])[e] = Bs[s][kc0 + (e & 7) + ((e >> 3) << 4)][n + (lane & 15)];
            }
        }
        #pragma unroll
        for (int tm = 0; tm < 4; ++tm) {
            #pragma unroll
            for (int tn = 0; tn < 2; ++tn) {
                acc[tm][tn] = __builtin_amdgcn_wmma_f32_16x16x32_bf16(
                    false, a[tm][0].v, false, b[tn][1].v, (short)0, acc[tm][tn], false, false);
                acc[tm][tn] = __builtin_amdgcn_wmma_f32_16x16x32_bf16(
                    false, a[tm][1].v, false, b[tn][0].v, (short)0, acc[tm][tn], false, false);
                acc[tm][tn] = __builtin_amdgcn_wmma_f32_16x16x32_bf16(
                    false, a[tm][0].v, false, b[tn][0].v, (short)0, acc[tm][tn], false, false);
            }
        }
    }

    #pragma unroll
    for (int tn = 0; tn < 2; ++tn) {
        int n = nBase + wn * 32 + tn * 16 + (lane & 15);
        float bv = Bias[n];
        #pragma unroll
        for (int tm = 0; tm < 4; ++tm) {
            int m0 = mBase + wm * 64 + tm * 16 + ((lane >> 4) << 3);
            #pragma unroll
            for (int r = 0; r < 8; ++r)
                Out[(size_t)(m0 + r) * NN + n] = acc[tm][tn][r] + bv;
        }
    }
}

extern "C" void kernel_launch(void* const* d_in, const int* in_sizes, int n_in,
                              void* d_out, int out_size, void* d_ws, size_t ws_size,
                              hipStream_t stream) {
    const float* x    = (const float*)d_in[0];
    const float* kern = (const float*)d_in[1];
    const float* wnd  = (const float*)d_in[2];
    const float* bias = (const float*)d_in[3];
    float* out = (float*)d_out;

    const size_t PLANE = (size_t)KK * NN;           // elements per plane
    const size_t NEED  = 4 * PLANE * sizeof(unsigned short); // 128 MB

    dim3 grid(NN / BN, MM / BM);
    dim3 block(THREADS);

    if (ws_size >= NEED) {
        unsigned short* Xhi = (unsigned short*)d_ws;
        unsigned short* Xlo = Xhi + PLANE;
        unsigned short* Phi = Xlo + PLANE;
        unsigned short* Plo = Phi + PLANE;
        szdense_split_prepass<<<dim3((unsigned)(PLANE / (256 * 4))), dim3(256), 0, stream>>>(
            x, kern, wnd, Xhi, Xlo, Phi, Plo);
        szdense_wmma_tdm<<<grid, block, 0, stream>>>(Xhi, Xlo, Phi, Plo, bias, out);
    } else {
        szdense_wmma_bf16x2<<<grid, block, 0, stream>>>(x, kern, wnd, bias, out);
    }
}
